// DPModel_36464272343523
// MI455X (gfx1250) — matrix operand order repaired and checked by reading, not compiled, not run on hardware
//
#include <hip/hip_runtime.h>
#include <hip/hip_bf16.h>

typedef __attribute__((ext_vector_type(16))) _Float16 v16h;
typedef __attribute__((ext_vector_type(8)))  float    v8f;

#define Y_ 2
#define NI_ 2048
#define MJ_ 64
#define N_ (Y_*NI_)   // 4096 atoms
#define M_ (Y_*MJ_)   // 128 neighbors
#define RCUT_ 6.0f
#define RS_ 3.0f

union U16h { _Float16 h[16]; v16h v; };

static __device__ inline v8f wmma_f16(v16h a, v16h b, v8f c) {
  // D = A(16x32 f16) * B(32x16 f16) + C(16x16 f32)
  return __builtin_amdgcn_wmma_f32_16x16x32_f16(false, a, false, b, (short)0, c, false, false);
}

// Branch-free tanh on native v_exp_f32 / v_rcp_f32 (TRANS ops co-execute with WMMA):
// tanh(x) = 1 - 2 / (exp2(2*log2e*x) + 1)
static __device__ inline float fast_tanh(float x) {
  float t = fminf(fmaxf(x, -15.0f), 15.0f);
  float e = __builtin_amdgcn_exp2f(t * 2.8853900817779268f);  // 2*log2(e)
  return 1.0f - 2.0f * __builtin_amdgcn_rcpf(e + 1.0f);
}

// ---------------------------------------------------------------------------
// Prep: f32 -> f16 weight conversion (done once per launch into workspace)
// ---------------------------------------------------------------------------
__global__ void cvt_f16_kernel(const float* __restrict__ src,
                               _Float16* __restrict__ dst, int n) {
  int idx = blockIdx.x * blockDim.x + threadIdx.x;
  if (idx < n) dst[idx] = (_Float16)src[idx];
}

// ---------------------------------------------------------------------------
// Stage 1: per-pair geometry -> sc (f32) and 10-channel RX padded to 16 (f16)
// ---------------------------------------------------------------------------
__global__ void geom_kernel(const float* __restrict__ coord,
                            const float* __restrict__ box,
                            const int* __restrict__ nbrs,
                            const float* __restrict__ sr_mean,
                            const float* __restrict__ sr_std,
                            float* __restrict__ sc_out,
                            _Float16* __restrict__ rx_out) {
  int pair = blockIdx.x * blockDim.x + threadIdx.x;
  if (pair >= N_ * M_) return;
  int n = pair >> 7;            // / 128
  int i = n / NI_;              // center type
  int jn = nbrs[pair];
  float dx = coord[0*N_ + jn] - coord[0*N_ + n];
  float dy = coord[1*N_ + jn] - coord[1*N_ + n];
  float dz = coord[2*N_ + jn] - coord[2*N_ + n];
  // general 3x3 box: minimum image d -= box @ round(inv(box) @ d)
  float m00=box[0],m01=box[1],m02=box[2];
  float m10=box[3],m11=box[4],m12=box[5];
  float m20=box[6],m21=box[7],m22=box[8];
  float det = m00*(m11*m22-m12*m21) - m01*(m10*m22-m12*m20) + m02*(m10*m21-m11*m20);
  float id = 1.0f / det;
  float f0 = ((m11*m22-m12*m21)*dx + (m02*m21-m01*m22)*dy + (m01*m12-m02*m11)*dz)*id;
  float f1 = ((m12*m20-m10*m22)*dx + (m00*m22-m02*m20)*dy + (m02*m10-m00*m12)*dz)*id;
  float f2 = ((m10*m21-m11*m20)*dx + (m01*m20-m00*m21)*dy + (m00*m11-m01*m10)*dz)*id;
  float r0 = rintf(f0), r1 = rintf(f1), r2 = rintf(f2);
  dx -= m00*r0 + m01*r1 + m02*r2;
  dy -= m10*r0 + m11*r1 + m12*r2;
  dz -= m20*r0 + m21*r1 + m22*r2;
  float r = sqrtf(dx*dx + dy*dy + dz*dz + 1e-18f);
  float u = (r - RS_) / (RCUT_ - RS_);
  float sw = (r < RS_) ? 1.0f
           : ((r < RCUT_) ? (((-6.0f*u + 15.0f)*u - 10.0f)*u*u*u + 1.0f) : 0.0f);
  float sr = (r > 1e-6f) ? sw / fmaxf(r, 1e-6f) : 0.0f;
  float inv_std = 1.0f / sr_std[i];
  float srn = sr * inv_std;
  float sc  = (sr - sr_mean[i]) * inv_std;
  float ir = 1.0f / (r + 1e-16f);
  float x0 = dx*ir, x1 = dy*ir, x2 = dz*ir;
  const float s3 = 1.7320508075688772f;
  const float s2 = 1.4142135623730951f;
  float rx[10];
  rx[0] = srn;
  rx[1] = s3*srn*x0; rx[2] = s3*srn*x1; rx[3] = s3*srn*x2;
  float t = 3.0f * srn;
  rx[4] = t*(x0*x0 - (1.0f/3.0f));
  rx[5] = t*s2*x0*x1;
  rx[6] = t*s2*x0*x2;
  rx[7] = t*(x1*x1 - (1.0f/3.0f));
  rx[8] = t*s2*x1*x2;
  rx[9] = t*(x2*x2 - (1.0f/3.0f));
  sc_out[pair] = sc;
  _Float16* o = rx_out + (size_t)pair * 16;
  #pragma unroll
  for (int c = 0; c < 10; ++c) o[c] = (_Float16)rx[c];
  #pragma unroll
  for (int c = 10; c < 16; ++c) o[c] = (_Float16)0.f;
}

// ---------------------------------------------------------------------------
// Stage 2: embedding MLP 1->32->64->64, 16 pairs/wave, WMMA f16.
// Weights staged once per block into LDS with coalesced b32 copies.
// ---------------------------------------------------------------------------
__global__ __launch_bounds__(128, 1)
void embed_kernel(const float* __restrict__ sc_g,
                  const float* __restrict__ eW1, const float* __restrict__ eb1,
                  const _Float16* __restrict__ w2h, const float* __restrict__ eb2,
                  const _Float16* __restrict__ w3h, const float* __restrict__ eb3,
                  _Float16* __restrict__ embed_out) {
  __shared__ alignas(16) _Float16 sW2[32 * 64];        // 4 KB
  __shared__ alignas(16) _Float16 sW3[64 * 64];        // 8 KB
  __shared__ alignas(16) _Float16 lds_h2[4][16 * 64];  // 8 KB transpose staging
  int lane = threadIdx.x & 31;
  int wave = threadIdx.x >> 5;
  int ij = blockIdx.y;
  int i = ij >> 1, j = ij & 1;
  int tile = blockIdx.x * 4 + wave;
  int p0 = tile * 16;

  int arow  = lane & 15;
  int kgrp8 = (lane >> 4) * 8;
  int bcol  = lane & 15;
  int bk16  = (lane >> 4) * 16;
  int rbase = (lane >> 4) * 8;

  // cooperative coalesced weight staging (f16, 2 halves per b32)
  {
    const unsigned* s2p = (const unsigned*)(w2h + ij * 2048);
    unsigned* d2p = (unsigned*)sW2;
    for (int t = threadIdx.x; t < 1024; t += 128) d2p[t] = s2p[t];
    const unsigned* s3p = (const unsigned*)(w3h + ij * 4096);
    unsigned* d3p = (unsigned*)sW3;
    for (int t = threadIdx.x; t < 2048; t += 128) d3p[t] = s3p[t];
  }
  __syncthreads();

  const float* w1  = eW1 + ij * 32;
  const float* b1  = eb1 + ij * 32;
  const float* b2v = eb2 + ij * 64;
  const float* b3v = eb3 + ij * 64;

  auto pair_of = [&](int r) {
    int p = p0 + r;
    int n = i * NI_ + (p >> 6);
    int m = j * MJ_ + (p & 63);
    return n * M_ + m;
  };

  // ---- layer 1: h1 (16x32) built directly in WMMA A layout ----
  float scA = sc_g[pair_of(arow)];
  U16h a1;
  #pragma unroll
  for (int e = 0; e < 16; ++e) {
    int k = kgrp8 + (e < 8 ? e : e + 8);      // 0..31
    a1.h[e] = (_Float16)fast_tanh(scA * w1[k] + b1[k]);
  }

  // ---- layer 2: 4 WMMAs ----
  v8f acc2[4];
  #pragma unroll
  for (int nt = 0; nt < 4; ++nt) {
    U16h b;
    #pragma unroll
    for (int e = 0; e < 16; ++e)
      b.h[e] = sW2[(kgrp8 + (e < 8 ? e : e + 8)) * 64 + nt * 16 + bcol];
    v8f z = {};
    acc2[nt] = wmma_f16(a1.v, b.v, z);
  }

  // ---- epilogue L2: h2 = tanh(acc+b2) + concat(h1,h1)[col], C layout ----
  float scrow[8];
  #pragma unroll
  for (int r = 0; r < 8; ++r) scrow[r] = sc_g[pair_of(rbase + r)];

  float h2c[4][8];
  #pragma unroll
  for (int nt = 0; nt < 4; ++nt) {
    int col = nt * 16 + bcol;
    float w1c = w1[col & 31];
    float b1c = b1[col & 31];
    float bb  = b2v[col];
    #pragma unroll
    for (int r = 0; r < 8; ++r) {
      float h1v = fast_tanh(scrow[r] * w1c + b1c);
      h2c[nt][r] = fast_tanh(acc2[nt][r] + bb) + h1v;
    }
  }

  // stage h2 to LDS row-major (C layout -> A layout transpose)
  #pragma unroll
  for (int nt = 0; nt < 4; ++nt) {
    int col = nt * 16 + bcol;
    #pragma unroll
    for (int r = 0; r < 8; ++r)
      lds_h2[wave][(rbase + r) * 64 + col] = (_Float16)h2c[nt][r];
  }
  __syncthreads();

  U16h a3[2];
  #pragma unroll
  for (int kc = 0; kc < 2; ++kc)
    #pragma unroll
    for (int e = 0; e < 16; ++e) {
      int k = kc * 32 + kgrp8 + (e < 8 ? e : e + 8);
      a3[kc].h[e] = lds_h2[wave][arow * 64 + k];
    }

  // ---- layer 3: 8 WMMAs ----
  v8f acc3[4];
  #pragma unroll
  for (int nt = 0; nt < 4; ++nt) {
    v8f acc = {};
    #pragma unroll
    for (int kc = 0; kc < 2; ++kc) {
      U16h b;
      #pragma unroll
      for (int e = 0; e < 16; ++e)
        b.h[e] = sW3[(kc * 32 + kgrp8 + (e < 8 ? e : e + 8)) * 64 + nt * 16 + bcol];
      acc = wmma_f16(a3[kc].v, b.v, acc);
    }
    acc3[nt] = acc;
  }

  // ---- epilogue L3: h3 = tanh(acc+b3) + h2 ; store f16 embed ----
  #pragma unroll
  for (int nt = 0; nt < 4; ++nt) {
    int col = nt * 16 + bcol;
    float bb = b3v[col];
    #pragma unroll
    for (int r = 0; r < 8; ++r) {
      float h3 = fast_tanh(acc3[nt][r] + bb) + h2c[nt][r];
      embed_out[(size_t)pair_of(rbase + r) * 64 + col] = (_Float16)h3;
    }
  }
}

// ---------------------------------------------------------------------------
// Stage 3: per atom T(16x64) = RXpad(16x128) @ embed(128x64) / 64.
// 4 atoms per block (wave per atom); RX/embed staged into LDS coalesced.
// ---------------------------------------------------------------------------
__global__ __launch_bounds__(128, 1)
void tmat_kernel(const _Float16* __restrict__ rx,
                 const _Float16* __restrict__ embed,
                 float* __restrict__ T_out) {
  __shared__ alignas(16) _Float16 sRX[4][128 * 16];  // 16 KB
  __shared__ alignas(16) _Float16 sEM[4][128 * 64];  // 64 KB
  int lane = threadIdx.x & 31;
  int wave = threadIdx.x >> 5;
  int n = blockIdx.x * 4 + wave;

  int arow  = lane & 15;
  int kgrp8 = (lane >> 4) * 8;
  int bcol  = lane & 15;
  int bk16  = (lane >> 4) * 16;
  int rbase = (lane >> 4) * 8;

  {
    const unsigned* rs = (const unsigned*)(rx + (size_t)n * M_ * 16);
    unsigned* rd = (unsigned*)sRX[wave];
    for (int t = lane; t < 1024; t += 32) rd[t] = rs[t];
    const unsigned* es = (const unsigned*)(embed + (size_t)n * M_ * 64);
    unsigned* ed = (unsigned*)sEM[wave];
    for (int t = lane; t < 4096; t += 32) ed[t] = es[t];
  }
  __syncthreads();

  v8f acc[4] = {};
  #pragma unroll
  for (int kc = 0; kc < 4; ++kc) {
    U16h a;
    #pragma unroll
    for (int e = 0; e < 16; ++e) {
      int m = kc * 32 + kgrp8 + (e < 8 ? e : e + 8);
      a.h[e] = sRX[wave][m * 16 + arow];
    }
    #pragma unroll
    for (int nt = 0; nt < 4; ++nt) {
      U16h b;
      #pragma unroll
      for (int e = 0; e < 16; ++e) {
        int m = kc * 32 + kgrp8 + (e < 8 ? e : e + 8);
        b.h[e] = sEM[wave][m * 64 + nt * 16 + bcol];
      }
      acc[nt] = wmma_f16(a.v, b.v, acc[nt]);
    }
  }
  float* Tn = T_out + (size_t)n * 1024;     // 16 channels x 64
  #pragma unroll
  for (int nt = 0; nt < 4; ++nt)
    #pragma unroll
    for (int r = 0; r < 8; ++r)
      Tn[(rbase + r) * 64 + nt * 16 + bcol] = acc[nt][r] * (1.0f / 64.0f);
}

// ---------------------------------------------------------------------------
// Stage 4a: per-atom G(4x64) assembly -> f16 Gf (N,256)
// ---------------------------------------------------------------------------
__global__ void gmat_kernel(const float* __restrict__ T_in,
                            const float* __restrict__ Tbias,
                            _Float16* __restrict__ G_out) {
  int n = blockIdx.x;
  int w = threadIdx.x;                       // 0..63
  const float* Tn = T_in + (size_t)n * 1024;
  const float s2 = 1.4142135623730951f;
  float T0w = Tn[w] + Tbias[w];
  #pragma unroll
  for (int a = 0; a < 4; ++a) {
    float T0a = Tn[a] + Tbias[a];
    float g = T0w * T0a;
    #pragma unroll
    for (int c = 0; c < 3; ++c)
      g += Tn[(1 + c) * 64 + w] * Tn[(1 + c) * 64 + a];
    float x0 = Tn[1 * 64 + 4 + a];
    float x1 = Tn[2 * 64 + 4 + a];
    float x2 = Tn[3 * 64 + 4 + a];
    float t6[6] = { x0*x0, s2*x0*x1, s2*x0*x2, x1*x1, s2*x1*x2, x2*x2 };
    #pragma unroll
    for (int c = 0; c < 6; ++c)
      g += (t6[c] + Tn[(4 + c) * 64 + 4 + a]) * Tn[(4 + c) * 64 + w];
    G_out[(size_t)n * 256 + a * 64 + w] = (_Float16)g;
  }
}

// ---------------------------------------------------------------------------
// Stage 4b: fitting layer [2048x256] @ [256x256] + bias, tanh, opt residual.
// X tile and per-k-chunk weight tile staged into LDS with coalesced copies.
// ---------------------------------------------------------------------------
__global__ __launch_bounds__(128, 1)
void fit_layer_kernel(const _Float16* __restrict__ X,
                      const _Float16* __restrict__ Wh,
                      const float* __restrict__ bias,
                      const _Float16* __restrict__ resid,
                      _Float16* __restrict__ out) {
  __shared__ alignas(16) _Float16 sX[16 * 256];   // 8 KB
  __shared__ alignas(16) _Float16 sW[32 * 256];   // 16 KB
  int lane = threadIdx.x & 31;
  int wave = threadIdx.x >> 5;
  int i = blockIdx.y;
  int rowtile = blockIdx.x;
  int colbase = wave * 64;

  int arow  = lane & 15;
  int kgrp8 = (lane >> 4) * 8;
  int bcol  = lane & 15;
  int bk16  = (lane >> 4) * 16;
  int rbase = (lane >> 4) * 8;

  const _Float16* Xi = X + ((size_t)i * NI_ + rowtile * 16) * 256;
  const _Float16* Wi = Wh + (size_t)i * 256 * 256;

  {
    const unsigned* xs = (const unsigned*)Xi;     // 16*256 halves = 2048 b32
    unsigned* xd = (unsigned*)sX;
    for (int t = threadIdx.x; t < 2048; t += 128) xd[t] = xs[t];
  }

  v8f acc[4] = {};
  for (int kc = 0; kc < 8; ++kc) {
    __syncthreads();   // previous tile consumed (and sX ready on first iter)
    {
      const unsigned* wsrc = (const unsigned*)(Wi + (size_t)kc * 32 * 256);
      unsigned* wdst = (unsigned*)sW;
      for (int t = threadIdx.x; t < 4096; t += 128) wdst[t] = wsrc[t];
    }
    __syncthreads();
    U16h a;
    #pragma unroll
    for (int e = 0; e < 16; ++e) {
      int k = kc * 32 + kgrp8 + (e < 8 ? e : e + 8);
      a.h[e] = sX[arow * 256 + k];
    }
    #pragma unroll
    for (int nt = 0; nt < 4; ++nt) {
      U16h b;
      #pragma unroll
      for (int e = 0; e < 16; ++e)
        b.h[e] = sW[(kgrp8 + (e < 8 ? e : e + 8)) * 256 + colbase + nt * 16 + bcol];
      acc[nt] = wmma_f16(a.v, b.v, acc[nt]);
    }
  }
  #pragma unroll
  for (int nt = 0; nt < 4; ++nt) {
    int col = colbase + nt * 16 + bcol;
    float bb = bias[i * 256 + col];
    #pragma unroll
    for (int r = 0; r < 8; ++r) {
      size_t idx = ((size_t)i * NI_ + rowtile * 16 + rbase + r) * 256 + col;
      float h = fast_tanh(acc[nt][r] + bb);
      if (resid) h += (float)resid[idx];
      out[idx] = (_Float16)h;
    }
  }
}

// ---------------------------------------------------------------------------
// Stage 5: energy head — wave-per-atom dot(h2, fWo) + fbo + Ebias, atomic sum
// ---------------------------------------------------------------------------
__global__ void zero_kernel(float* out) { out[0] = 0.0f; }

__global__ void energy_kernel(const _Float16* __restrict__ H2,
                              const float* __restrict__ fWo,
                              const float* __restrict__ fbo,
                              const float* __restrict__ Ebias,
                              float* __restrict__ out) {
  int lane = threadIdx.x & 31;
  int wave = threadIdx.x >> 5;
  int n = blockIdx.x * 8 + wave;
  int i = n / NI_;
  const _Float16* h = H2 + (size_t)n * 256;
  const float* wo = fWo + i * 256;
  float s = 0.f;
  #pragma unroll
  for (int k = lane; k < 256; k += 32) s += (float)h[k] * wo[k];
  #pragma unroll
  for (int o = 16; o > 0; o >>= 1) s += __shfl_xor(s, o, 32);
  if (lane == 0) atomicAdd(out, s + fbo[i] + Ebias[i]);
}

// ---------------------------------------------------------------------------
extern "C" void kernel_launch(void* const* d_in, const int* in_sizes, int n_in,
                              void* d_out, int out_size, void* d_ws, size_t ws_size,
                              hipStream_t stream) {
  (void)in_sizes; (void)n_in; (void)out_size; (void)ws_size;
  const float* coord   = (const float*)d_in[0];
  const float* box     = (const float*)d_in[1];
  const int*   nbrs    = (const int*)  d_in[2];
  const float* sr_mean = (const float*)d_in[3];
  const float* sr_std  = (const float*)d_in[4];
  const float* eW1     = (const float*)d_in[5];
  const float* eb1     = (const float*)d_in[6];
  const float* eW2     = (const float*)d_in[7];
  const float* eb2     = (const float*)d_in[8];
  const float* eW3     = (const float*)d_in[9];
  const float* eb3     = (const float*)d_in[10];
  const float* Tbias   = (const float*)d_in[11];
  const float* fW1     = (const float*)d_in[12];
  const float* fb1     = (const float*)d_in[13];
  const float* fW2     = (const float*)d_in[14];
  const float* fb2     = (const float*)d_in[15];
  const float* fWo     = (const float*)d_in[16];
  const float* fbo     = (const float*)d_in[17];
  const float* Ebias   = (const float*)d_in[18];
  float* outf = (float*)d_out;

  char* ws = (char*)d_ws;
  size_t off = 0;
  float*    sc    = (float*)   (ws + off); off += (size_t)N_ * M_ * sizeof(float);
  _Float16* rx    = (_Float16*)(ws + off); off += (size_t)N_ * M_ * 16 * sizeof(_Float16);
  _Float16* embed = (_Float16*)(ws + off); off += (size_t)N_ * M_ * 64 * sizeof(_Float16);
  float*    T     = (float*)   (ws + off); off += (size_t)N_ * 1024 * sizeof(float);
  _Float16* G     = (_Float16*)(ws + off); off += (size_t)N_ * 256 * sizeof(_Float16);
  _Float16* H1    = (_Float16*)(ws + off); off += (size_t)N_ * 256 * sizeof(_Float16);
  _Float16* H2    = (_Float16*)(ws + off); off += (size_t)N_ * 256 * sizeof(_Float16);
  _Float16* w2h   = (_Float16*)(ws + off); off += (size_t)8192   * sizeof(_Float16);
  _Float16* w3h   = (_Float16*)(ws + off); off += (size_t)16384  * sizeof(_Float16);
  _Float16* fW1h  = (_Float16*)(ws + off); off += (size_t)131072 * sizeof(_Float16);
  _Float16* fW2h  = (_Float16*)(ws + off); off += (size_t)131072 * sizeof(_Float16);

  // weight pre-conversion to f16
  cvt_f16_kernel<<<dim3(32),  dim3(256), 0, stream>>>(eW2, w2h, 8192);
  cvt_f16_kernel<<<dim3(64),  dim3(256), 0, stream>>>(eW3, w3h, 16384);
  cvt_f16_kernel<<<dim3(512), dim3(256), 0, stream>>>(fW1, fW1h, 131072);
  cvt_f16_kernel<<<dim3(512), dim3(256), 0, stream>>>(fW2, fW2h, 131072);

  int npair = N_ * M_;
  geom_kernel<<<dim3((npair + 255) / 256), dim3(256), 0, stream>>>(
      coord, box, nbrs, sr_mean, sr_std, sc, rx);
  embed_kernel<<<dim3(2048, 4), dim3(128), 0, stream>>>(
      sc, eW1, eb1, w2h, eb2, w3h, eb3, embed);
  tmat_kernel<<<dim3(N_ / 4), dim3(128), 0, stream>>>(rx, embed, T);
  gmat_kernel<<<dim3(N_), dim3(64), 0, stream>>>(T, Tbias, G);
  fit_layer_kernel<<<dim3(128, 2), dim3(128), 0, stream>>>(G, fW1h, fb1, nullptr, H1);
  fit_layer_kernel<<<dim3(128, 2), dim3(128), 0, stream>>>(H1, fW2h, fb2, H1, H2);
  zero_kernel<<<dim3(1), dim3(1), 0, stream>>>(outf);
  energy_kernel<<<dim3(512), dim3(256), 0, stream>>>(H2, fWo, fbo, Ebias, outf);
}
